// SAC_Attention_Critic_33535104647964
// MI455X (gfx1250) — matrix operand order repaired and checked by reading, not compiled, not run on hardware
//
#include <hip/hip_runtime.h>

// ---------------- Types / WMMA fragment helpers (gfx1250, wave32) ----------------
typedef __attribute__((ext_vector_type(16))) __bf16 bf16x16;
typedef __attribute__((ext_vector_type(8)))  __bf16 bf16x8;
typedef __attribute__((ext_vector_type(8)))  float  f32x8;

union ABFrag { bf16x16 v; struct { bf16x8 lo, hi; } h; };

__device__ __forceinline__ f32x8 wmma_bf16(bf16x16 a, bf16x16 b, f32x8 c) {
  // D = A(16x32 bf16) * B(32x16 bf16) + C(16x16 f32)
  return __builtin_amdgcn_wmma_f32_16x16x32_bf16(false, a, false, b, (short)0, c, false, false);
}

// A-frag: row-major [16][K] source; lane&15 = row; half-wave selects K 0-7/16-23 vs 8-15/24-31
__device__ __forceinline__ bf16x16 afrag(const __bf16* base, int rowStride, int k0) {
  int lane = threadIdx.x & 31;
  int grp = lane >> 4, r = lane & 15;
  ABFrag f;
  f.h.lo = *(const bf16x8*)(base + r * rowStride + k0 + grp * 8);
  f.h.hi = *(const bf16x8*)(base + r * rowStride + k0 + 16 + grp * 8);
  return f.v;
}

// B-frag: column-major [N][K] source; lane&15 = column; half-wave selects K 0-7/16-23 vs 8-15/24-31
__device__ __forceinline__ bf16x16 bfrag(const __bf16* colBase, int colStride, int k0) {
  int lane = threadIdx.x & 31;
  int grp = lane >> 4, c = lane & 15;
  ABFrag f;
  f.h.lo = *(const bf16x8*)(colBase + c * colStride + k0 + grp * 8);
  f.h.hi = *(const bf16x8*)(colBase + c * colStride + k0 + 16 + grp * 8);
  return f.v;
}

__device__ __forceinline__ float lrelu(float x) { return x > 0.f ? x : 0.01f * x; }

// ---------------- Problem constants ----------------
#define NA 16
#define NB 8192
#define NO 128
#define NACT 16
#define NH_ 128
#define NIN 144
#define NINP 160   // padded to 5 K-tiles
#define NCH 64     // batch chunks for BN stats

// ---------------- Workspace layout (bytes) ----------------
constexpr size_t OFF_PART = 0;                           // [16][64][144][2] f32
constexpr size_t OFF_MEAN = OFF_PART + (size_t)NA*NCH*NIN*2*4;
constexpr size_t OFF_RSTD = OFF_MEAN + (size_t)NA*NIN*4;
constexpr size_t OFF_WSA  = OFF_RSTD + (size_t)NA*NIN*4;        // [16][128][160] bf16
constexpr size_t OFF_WSE  = OFF_WSA + (size_t)NA*128*NINP*2;    // [16][128][128] bf16
constexpr size_t OFF_WK   = OFF_WSE + (size_t)NA*128*128*2;     // [128][128] bf16
constexpr size_t OFF_WSEL = OFF_WK  + (size_t)128*128*2;
constexpr size_t OFF_WV   = OFF_WSEL+ (size_t)128*128*2;
constexpr size_t OFF_WC1  = OFF_WV  + (size_t)128*128*2;        // [16][128][256] bf16
constexpr size_t OFF_WC2  = OFF_WC1 + (size_t)NA*128*256*2;     // [16][16][128] bf16

// ---------------- Kernel 1: BN partial sums (deterministic, no atomics) ----------------
__global__ void stats_partial(const float* __restrict__ obs, const float* __restrict__ act,
                              float* __restrict__ part) {
  int a = blockIdx.x / NCH, ch = blockIdx.x % NCH;
  int f = threadIdx.x;
  if (f >= NIN) return;
  float s = 0.f, sq = 0.f;
  int b0 = ch * (NB / NCH);
  for (int r = 0; r < NB / NCH; ++r) {
    int b = b0 + r;
    float x = (f < NO) ? obs[((size_t)a * NB + b) * NO + f]
                       : act[((size_t)a * NB + b) * NACT + (f - NO)];
    s += x; sq += x * x;
  }
  part[(((size_t)a * NCH + ch) * NIN + f) * 2 + 0] = s;
  part[(((size_t)a * NCH + ch) * NIN + f) * 2 + 1] = sq;
}

__global__ void stats_final(const float* __restrict__ part, float* __restrict__ mean,
                            float* __restrict__ rstd) {
  int i = blockIdx.x * blockDim.x + threadIdx.x;
  if (i >= NA * NIN) return;
  int a = i / NIN, f = i % NIN;
  float s = 0.f, sq = 0.f;
  for (int ch = 0; ch < NCH; ++ch) {
    s  += part[(((size_t)a * NCH + ch) * NIN + f) * 2 + 0];
    sq += part[(((size_t)a * NCH + ch) * NIN + f) * 2 + 1];
  }
  float m = s / (float)NB;
  float v = sq / (float)NB - m * m;
  mean[i] = m;
  rstd[i] = rsqrtf(v + 1e-5f);
}

// ---------------- Kernel 2: weights -> bf16, transposed to [N][K] (B-frag friendly) ----------------
__global__ void prep_weights(const float* __restrict__ W_sa, const float* __restrict__ W_s,
                             const float* __restrict__ Wk, const float* __restrict__ Wsel,
                             const float* __restrict__ Wv, const float* __restrict__ Wc1,
                             const float* __restrict__ Wc2,
                             __bf16* __restrict__ t_sa, __bf16* __restrict__ t_s,
                             __bf16* __restrict__ t_k, __bf16* __restrict__ t_sel,
                             __bf16* __restrict__ t_v, __bf16* __restrict__ t_c1,
                             __bf16* __restrict__ t_c2) {
  long i = (long)blockIdx.x * blockDim.x + threadIdx.x;
  const long N0 = (long)NA * 128 * NINP;   // Wt_sa
  const long N1 = (long)NA * 128 * 128;    // Wt_s
  const long N2 = 128L * 128;              // each of k/sel/v
  const long N5 = (long)NA * 128 * 256;    // Wt_c1
  const long N6 = (long)NA * 16 * 128;     // Wt_c2
  if (i < N0) {
    int a = (int)(i / (128 * NINP)); int r = (int)(i % (128 * NINP));
    int n = r / NINP, k = r % NINP;
    t_sa[i] = (__bf16)((k < NIN) ? W_sa[((size_t)a * NIN + k) * 128 + n] : 0.f);
    return;
  } i -= N0;
  if (i < N1) {
    int a = (int)(i / (128 * 128)); int r = (int)(i % (128 * 128));
    int n = r / 128, k = r % 128;
    t_s[i] = (__bf16)W_s[((size_t)a * 128 + k) * 128 + n];
    return;
  } i -= N1;
  if (i < 3 * N2) {
    int which = (int)(i / N2); long j = i % N2;
    int c = (int)(j / 128), k = (int)(j % 128);           // c = nh*32 + d
    size_t src = (((size_t)(c >> 5)) * 128 + k) * 32 + (c & 31);
    __bf16* dst = (which == 0) ? t_k : (which == 1) ? t_sel : t_v;
    const float* s = (which == 0) ? Wk : (which == 1) ? Wsel : Wv;
    dst[j] = (__bf16)s[src];
    return;
  } i -= 3 * N2;
  if (i < N5) {
    int a = (int)(i / (128 * 256)); int r = (int)(i % (128 * 256));
    int n = r / 256, k = r % 256;
    t_c1[i] = (__bf16)Wc1[((size_t)a * 256 + k) * 128 + n];
    return;
  } i -= N5;
  if (i < N6) {
    int a = (int)(i / (16 * 128)); int r = (int)(i % (16 * 128));
    int n = r / 128, k = r % 128;
    t_c2[i] = (__bf16)Wc2[((size_t)a * 128 + k) * 16 + n];
  }
}

// ---------------- Kernel 3: fully fused forward ----------------
// Block: 512 threads = 16 wave32s; wave w owns agent w for per-agent GEMMs.
// Batch tile = 16 rows (one WMMA M-tile); grid = B/16 = 512 blocks.
__global__ __launch_bounds__(512)
void fused_main(const float* __restrict__ obs, const float* __restrict__ actions,
                const float* __restrict__ b_sa, const float* __restrict__ b_s,
                const float* __restrict__ bv, const float* __restrict__ bc1,
                const float* __restrict__ bc2,
                const float* __restrict__ mean, const float* __restrict__ rstd,
                const __bf16* __restrict__ t_sa, const __bf16* __restrict__ t_s,
                const __bf16* __restrict__ t_k, const __bf16* __restrict__ t_sel,
                const __bf16* __restrict__ t_v, const __bf16* __restrict__ t_c1,
                const __bf16* __restrict__ t_c2,
                float* __restrict__ out) {
  // LDS: 65536*3 + 98304 + 16384 + 32 = 311,328 B of the WGP's 320 KB
  __shared__ __align__(16) __bf16 sa_s[16][16][128];      // [agent][row][h]
  __shared__ __align__(16) __bf16 se_s[16][16][128];      // [agent][row][h]
  __shared__ __align__(16) __bf16 other_s[16][16][128];   // [row][agent][nh*32+d]
  __shared__ __align__(16) __bf16 regA[49152];            // phased union (96 KB)
  __shared__ __align__(16) __bf16 w_s[2][16][16][16];     // [nl][row][i][j]
  __shared__ __align__(16) __bf16 zero16[16];

  const int tid  = threadIdx.x;
  const int wave = tid >> 5;      // = agent
  const int lane = tid & 31;
  const int grp  = lane >> 4;
  const int ln   = lane & 15;
  const int b0   = blockIdx.x * 16;
  const int a    = wave;

  // ---- Phase 0: load + batchnorm inputs -> regA as inpn[a][row][160] (bf16, zero-padded) ----
  if (tid < 16) zero16[tid] = (__bf16)0.f;
  for (int idx = tid; idx < 16 * 16 * NINP; idx += 512) {
    int aa = idx / (16 * NINP);
    int rr = (idx / NINP) & 15;
    int k  = idx % NINP;
    float xn = 0.f;
    if (k < NIN) {
      int b = b0 + rr;
      float x = (k < NO) ? obs[((size_t)aa * NB + b) * NO + k]
                         : actions[((size_t)aa * NB + b) * NACT + (k - NO)];
      xn = (x - mean[aa * NIN + k]) * rstd[aa * NIN + k];
    }
    regA[idx] = (__bf16)xn;
  }
  __syncthreads();

  // ---- Phase 1: sa = lrelu(bn(inp)@W_sa+b), se = lrelu(bn(obs)@W_s+b) ----
  {
    bf16x16 afr[5];
#pragma unroll
    for (int kt = 0; kt < 5; ++kt) afr[kt] = afrag(&regA[a * 16 * NINP], NINP, kt * 32);
#pragma unroll
    for (int nt = 0; nt < 8; ++nt) {
      f32x8 acc = {0, 0, 0, 0, 0, 0, 0, 0};
#pragma unroll
      for (int kt = 0; kt < 5; ++kt)
        acc = wmma_bf16(afr[kt], bfrag(t_sa + ((size_t)a * 128 + nt * 16) * NINP, NINP, kt * 32), acc);
      float bias = b_sa[a * 128 + nt * 16 + ln];
#pragma unroll
      for (int v = 0; v < 8; ++v)
        sa_s[a][v + grp * 8][nt * 16 + ln] = (__bf16)lrelu(acc[v] + bias);
    }
#pragma unroll
    for (int nt = 0; nt < 8; ++nt) {
      f32x8 acc = {0, 0, 0, 0, 0, 0, 0, 0};
#pragma unroll
      for (int kt = 0; kt < 4; ++kt)
        acc = wmma_bf16(afr[kt], bfrag(t_s + ((size_t)a * 128 + nt * 16) * 128, 128, kt * 32), acc);
      float bias = b_s[a * 128 + nt * 16 + ln];
#pragma unroll
      for (int v = 0; v < 8; ++v)
        se_s[a][v + grp * 8][nt * 16 + ln] = (__bf16)lrelu(acc[v] + bias);
    }
  }
  __syncthreads();

  // ---- Phase 2: heads in 2 groups of 2 (LDS reuse of regA) ----
  for (int hg = 0; hg < 2; ++hg) {
    __bf16* keys = regA;              // [row][nl][j][32d]   (col-major [j][d] per (row,nl))
    __bf16* selg = regA + 16384;      // [row][i][64]
    __bf16* valt = regA + 32768;      // [row][nl][32d][16j] (col-major [d][j] per (row,nl))

    // 2a: per-agent key/sel/val projections for this head group's 64 columns
    {
      bf16x16 asa[4], ase[4];
#pragma unroll
      for (int kt = 0; kt < 4; ++kt) {
        asa[kt] = afrag(&sa_s[a][0][0], 128, kt * 32);
        ase[kt] = afrag(&se_s[a][0][0], 128, kt * 32);
      }
#pragma unroll
      for (int ct = 0; ct < 4; ++ct) {
        int c0 = hg * 64 + ct * 16;
        int nl = ct >> 1;
        int dd = (ct & 1) * 16 + ln;
        f32x8 ck = {0,0,0,0,0,0,0,0}, cs = {0,0,0,0,0,0,0,0}, cv = {0,0,0,0,0,0,0,0};
#pragma unroll
        for (int kt = 0; kt < 4; ++kt) {
          ck = wmma_bf16(asa[kt], bfrag(t_k   + (size_t)c0 * 128, 128, kt * 32), ck);
          cs = wmma_bf16(ase[kt], bfrag(t_sel + (size_t)c0 * 128, 128, kt * 32), cs);
          cv = wmma_bf16(asa[kt], bfrag(t_v   + (size_t)c0 * 128, 128, kt * 32), cv);
        }
        float bvv = bv[c0 + ln];
#pragma unroll
        for (int v = 0; v < 8; ++v) {
          int b = v + grp * 8;
          keys[((b * 2 + nl) * 16 + a) * 32 + dd] = (__bf16)ck[v];
          selg[(b * 16 + a) * 64 + ct * 16 + ln]  = (__bf16)cs[v];
          valt[((b * 2 + nl) * 32 + dd) * 16 + a] = (__bf16)lrelu(cv[v] + bvv);
        }
      }
    }
    __syncthreads();

    // 2b/2c: attention; each wave handles 2 (head,row) pairs of the 32 in this group
#pragma unroll
    for (int q = 0; q < 2; ++q) {
      int p  = wave * 2 + q;
      int nl = p >> 4;
      int b  = p & 15;
      // logits[i][j] = sel(i,:) . keys(j,:)  -> one 16x16x32 WMMA
      bf16x16 aS = afrag(&selg[b * 16 * 64], 64, nl * 32);
      bf16x16 bK = bfrag(&keys[(b * 2 + nl) * 16 * 32], 32, 0);
      f32x8 lg = {0, 0, 0, 0, 0, 0, 0, 0};
      lg = wmma_bf16(aS, bK, lg);
      const float scale = 0.17677669529663687f;  // 1/sqrt(32)
#pragma unroll
      for (int v = 0; v < 8; ++v) {
        int i = v + grp * 8;
        float x = (ln == i) ? -1e9f : lg[v] * scale;
        float mx = x;
#pragma unroll
        for (int m = 1; m < 16; m <<= 1) mx = fmaxf(mx, __shfl_xor(mx, m, 16));
        float e = __expf(x - mx);
        float s = e;
#pragma unroll
        for (int m = 1; m < 16; m <<= 1) s += __shfl_xor(s, m, 16);
        w_s[nl][b][i][ln] = (__bf16)(e / s);
      }
      // other[i][d] = sum_j w[i][j] * vals[j][d]; K padded 16->32 with zeros
      ABFrag aw;
      aw.h.lo = *(const bf16x8*)(&w_s[nl][b][ln][grp * 8]);
      aw.h.hi = *(const bf16x8*)zero16;
#pragma unroll
      for (int dh = 0; dh < 2; ++dh) {
        ABFrag bvf;
        const __bf16* basev = &valt[((b * 2 + nl) * 32 + dh * 16 + ln) * 16];
        bvf.h.lo = *(const bf16x8*)(basev + grp * 8);
        bvf.h.hi = *(const bf16x8*)zero16;
        f32x8 oc = {0, 0, 0, 0, 0, 0, 0, 0};
        oc = wmma_bf16(aw.v, bvf.v, oc);
        int cb = (hg * 2 + nl) * 32 + dh * 16 + ln;
#pragma unroll
        for (int v = 0; v < 8; ++v)
          other_s[b][v + grp * 8][cb] = (__bf16)oc[v];
      }
    }
    __syncthreads();
  }

  // ---- Phase 3: critic MLP. cin = [se | other] (K=256), then H->ACT ----
  __bf16* h_s = regA;  // [agent][row][128] (safe: regA contents dead after last barrier)
  {
    bf16x16 afk[8];
#pragma unroll
    for (int kt = 0; kt < 4; ++kt) afk[kt] = afrag(&se_s[a][0][0], 128, kt * 32);
#pragma unroll
    for (int kt = 4; kt < 8; ++kt) afk[kt] = afrag(&other_s[0][a][0], 16 * 128, (kt - 4) * 32);
#pragma unroll
    for (int nt = 0; nt < 8; ++nt) {
      f32x8 acc = {0, 0, 0, 0, 0, 0, 0, 0};
#pragma unroll
      for (int kt = 0; kt < 8; ++kt)
        acc = wmma_bf16(afk[kt], bfrag(t_c1 + ((size_t)a * 128 + nt * 16) * 256, 256, kt * 32), acc);
      float bias = bc1[a * 128 + nt * 16 + ln];
#pragma unroll
      for (int v = 0; v < 8; ++v)
        h_s[((size_t)a * 16 + v + grp * 8) * 128 + nt * 16 + ln] = (__bf16)lrelu(acc[v] + bias);
    }
  }
  {
    f32x8 acc = {0, 0, 0, 0, 0, 0, 0, 0};
#pragma unroll
    for (int kt = 0; kt < 4; ++kt)
      acc = wmma_bf16(afrag(&h_s[(size_t)a * 16 * 128], 128, kt * 32),
                      bfrag(t_c2 + (size_t)a * 16 * 128, 128, kt * 32), acc);
    // q = all_q[argmax(actions)] ; C-frag: row = v + 8*grp, act = ln
#pragma unroll
    for (int v = 0; v < 8; ++v) {
      int r = v + grp * 8;
      int row = b0 + r;
      float qv = acc[v] + bc2[a * NACT + ln];
      float av = actions[((size_t)a * NB + row) * NACT + ln];
      int ai = ln;
#pragma unroll
      for (int m = 1; m < 16; m <<= 1) {
        float ov = __shfl_xor(av, m, 16);
        int   oi = __shfl_xor(ai, m, 16);
        if (ov > av || (ov == av && oi < ai)) { av = ov; ai = oi; }
      }
      float qsel = __shfl(qv, ai, 16);
      if (ln == 0) out[(size_t)a * NB + row] = qsel;
    }
  }
}

// ---------------- Host launcher ----------------
extern "C" void kernel_launch(void* const* d_in, const int* in_sizes, int n_in,
                              void* d_out, int out_size, void* d_ws, size_t ws_size,
                              hipStream_t stream) {
  (void)in_sizes; (void)n_in; (void)out_size; (void)ws_size;
  const float* obs     = (const float*)d_in[0];
  const float* actions = (const float*)d_in[1];
  const float* W_sa    = (const float*)d_in[2];
  const float* b_sa    = (const float*)d_in[3];
  const float* W_s     = (const float*)d_in[4];
  const float* b_s     = (const float*)d_in[5];
  const float* Wk      = (const float*)d_in[6];
  const float* Wsel    = (const float*)d_in[7];
  const float* Wv      = (const float*)d_in[8];
  const float* bv      = (const float*)d_in[9];
  const float* Wc1     = (const float*)d_in[10];
  const float* bc1     = (const float*)d_in[11];
  const float* Wc2     = (const float*)d_in[12];
  const float* bc2     = (const float*)d_in[13];

  char* ws = (char*)d_ws;
  float*  part = (float*)(ws + OFF_PART);
  float*  mean = (float*)(ws + OFF_MEAN);
  float*  rstd = (float*)(ws + OFF_RSTD);
  __bf16* t_sa = (__bf16*)(ws + OFF_WSA);
  __bf16* t_s  = (__bf16*)(ws + OFF_WSE);
  __bf16* t_k  = (__bf16*)(ws + OFF_WK);
  __bf16* t_sel= (__bf16*)(ws + OFF_WSEL);
  __bf16* t_v  = (__bf16*)(ws + OFF_WV);
  __bf16* t_c1 = (__bf16*)(ws + OFF_WC1);
  __bf16* t_c2 = (__bf16*)(ws + OFF_WC2);

  stats_partial<<<NA * NCH, 160, 0, stream>>>(obs, actions, part);
  stats_final<<<(NA * NIN + 255) / 256, 256, 0, stream>>>(part, mean, rstd);

  long prepN = (long)NA*128*NINP + (long)NA*128*128 + 3L*128*128 + (long)NA*128*256 + (long)NA*16*128;
  prep_weights<<<(int)((prepN + 255) / 256), 256, 0, stream>>>(
      W_sa, W_s, Wk, Wsel, Wv, Wc1, Wc2, t_sa, t_s, t_k, t_sel, t_v, t_c1, t_c2);

  fused_main<<<NB / 16, 512, 0, stream>>>(obs, actions, b_sa, b_s, bv, bc1, bc2,
                                          mean, rstd, t_sa, t_s, t_k, t_sel, t_v,
                                          t_c1, t_c2, (float*)d_out);
}
